// CapsuleNet_71811853189948
// MI455X (gfx1250) — matrix operand order repaired and checked by reading, not compiled, not run on hardware
//
#include <hip/hip_runtime.h>

// ---------------------------------------------------------------------------
// CDNA5 / gfx1250 CapsuleNet forward. wave32. bf16 WMMA (f32 accumulate) for
// the primary-capsule implicit GEMM (LDS double-buffered B panel) and the 3
// decoder GEMMs; VALU+LDS for conv1, squash, dynamic routing, masking.
// ---------------------------------------------------------------------------

typedef __attribute__((ext_vector_type(16))) __bf16 v16bf;
typedef __attribute__((ext_vector_type(8)))  __bf16 v8bf;
typedef __attribute__((ext_vector_type(8)))  float  v8f;

#define BATCH 512
#define NUM_CLASSES 10

// ---------------- fragment helpers (contiguous k order, k = 16h + s) --------------------

// A-matrix 16x32 bf16: lane<16 -> row m=lane, k = {0..15}
//                      lane>=16 -> row m=lane-16, k = {16..31}
__device__ __forceinline__ v16bf load_a_frag(const __bf16* __restrict__ row, int khalf) {
  v8bf lo = *(const v8bf*)(row + 2 * khalf);
  v8bf hi = *(const v8bf*)(row + 2 * khalf + 8);
  v16bf r;
#pragma unroll
  for (int e = 0; e < 8; ++e) { r[e] = lo[e]; r[e + 8] = hi[e]; }
  return r;
}

// ---------------- conv1: [512,1,28,28] -> h_t[b][20][20][256] bf16 ----------

__global__ void conv1_kernel(const float* __restrict__ x, const float* __restrict__ w,
                             const float* __restrict__ bias, __bf16* __restrict__ h_t) {
  int blk = blockIdx.x;               // (b, oy, ox)
  int ox = blk % 20; int t = blk / 20;
  int oy = t % 20;   int b = t / 20;
  int m  = threadIdx.x;               // 256 output channels
  __shared__ float xs[81];
  const float* xp = x + (size_t)b * 784 + oy * 28 + ox;
  if (threadIdx.x < 81) {
    int ky = threadIdx.x / 9, kx = threadIdx.x % 9;
    xs[threadIdx.x] = xp[ky * 28 + kx];
  }
  __syncthreads();
  const float* wp = w + (size_t)m * 81;
  float acc = bias[m];
#pragma unroll 9
  for (int k = 0; k < 81; ++k) acc = fmaf(xs[k], wp[k], acc);
  acc = fmaxf(acc, 0.f);
  h_t[(((size_t)b * 20 + oy) * 20 + ox) * 256 + m] = (__bf16)acc;
}

// ---- pack prim_w [256(m),256(c),9,9] f32 -> wA[kk][m][c] bf16 (K-contig) ---

__global__ void prep_wA(const float* __restrict__ prim_w, __bf16* __restrict__ wA, int n) {
  int i = blockIdx.x * 256 + threadIdx.x;
  if (i >= n) return;
  int c = i & 255; int t = i >> 8; int m = t & 255; int kk = t >> 8;
  wA[i] = (__bf16)prim_w[((size_t)m * 256 + c) * 81 + kk];
}

__global__ void f32_to_bf16(const float* __restrict__ in, __bf16* __restrict__ out, int n) {
  int i = blockIdx.x * 256 + threadIdx.x;
  if (i < n) out[i] = (__bf16)in[i];
}

__global__ void fill_zero(float* __restrict__ p, int n) {
  int i = blockIdx.x * 256 + threadIdx.x;
  if (i < n) p[i] = 0.f;
}

// ------- primary caps implicit GEMM: M=256, N=18432, K=81*256 (WMMA) -------
// Block tile 256x64 (8 waves, each 64x32); B panel staged in LDS, double-
// buffered so global B traffic is shared by all 8 waves. Output scattered
// into u[b][r=j*36+s][g] f32 (+bias).

__global__ void prim_gemm(const __bf16* __restrict__ hT, const __bf16* __restrict__ wA,
                          const float* __restrict__ pbias, float* __restrict__ u) {
  __shared__ __attribute__((aligned(32))) __bf16 Bs[2][64 * 256];   // 64 KB

  const int tid   = threadIdx.x;
  const int lane  = tid & 31;
  const int w     = tid >> 5;          // 8 waves
  const int mbase = (w & 3) * 64;      // 4 m-tiles of 64
  const int nloc  = (w >> 2) * 32;     // 2 n-halves of 32
  const int nbase = blockIdx.x * 64;   // 288 blocks cover N=18432

  const int khalfA = (lane < 16) ? 0 : 8;
  const int khalfB = (lane < 16) ? 0 : 16;

  // staging mapping: thread -> (column, channel-quarter); 128B contiguous each
  const int scol = tid & 63;
  const int sq   = tid >> 6;
  const int sn   = nbase + scol;
  const int sb   = sn / 36, sspos = sn % 36;
  const int soy  = sspos / 6, sox = sspos % 6;
  __bf16* dst0 = &Bs[0][scol * 256 + sq * 64];
  __bf16* dst1 = &Bs[1][scol * 256 + sq * 64];

#define STAGE(dst, ky, kx)                                                          \
  do {                                                                              \
    const __bf16* src = hT + (((size_t)sb * 20 + (2 * soy + (ky))) * 20 +           \
                              (2 * sox + (kx))) * 256 + sq * 64;                    \
    _Pragma("unroll")                                                               \
    for (int ch = 0; ch < 8; ++ch)                                                  \
      *(uint4*)((dst) + ch * 8) = *(const uint4*)(src + ch * 8);                    \
  } while (0)

  v8f zero = {0.f, 0.f, 0.f, 0.f, 0.f, 0.f, 0.f, 0.f};
  v8f acc[8] = {zero, zero, zero, zero, zero, zero, zero, zero};

  STAGE(dst0, 0, 0);                       // prologue: tap kk=0 into buffer 0

  int buf = 0;
  for (int kk = 0; kk < 81; ++kk) {
    __syncthreads();                       // staged buffer visible to all waves
    if (kk < 80) {                         // stage next tap into the other buffer
      int ky = (kk + 1) / 9, kx = (kk + 1) % 9;
      STAGE(buf ? dst0 : dst1, ky, kx);
    }
    const __bf16* arow = wA + (size_t)kk * 65536;
    if (kk < 80) __builtin_prefetch(arow + 65536, 0, 0);
    const __bf16* bbuf = Bs[buf];
#pragma unroll 2
    for (int c0 = 0; c0 < 256; c0 += 32) {
      v16bf bf0 = *(const v16bf*)(bbuf + (nloc +      (lane & 15)) * 256 + c0 + khalfB);
      v16bf bf1 = *(const v16bf*)(bbuf + (nloc + 16 + (lane & 15)) * 256 + c0 + khalfB);
#pragma unroll
      for (int f = 0; f < 4; ++f) {
        v16bf af = load_a_frag(arow + (size_t)(mbase + f * 16 + (lane & 15)) * 256 + c0,
                               khalfA);
        acc[f]     = __builtin_amdgcn_wmma_f32_16x16x32_bf16(false, af, false, bf0,
                                                             (short)0, acc[f], false, false);
        acc[f + 4] = __builtin_amdgcn_wmma_f32_16x16x32_bf16(false, af, false, bf1,
                                                             (short)0, acc[f + 4], false, false);
      }
    }
    buf ^= 1;
  }
#undef STAGE

  const int mrow = (lane < 16) ? 0 : 8;
#pragma unroll
  for (int nsub = 0; nsub < 2; ++nsub) {
    int n = nbase + nloc + nsub * 16 + (lane & 15);
    int b = n / 36, s = n % 36;
#pragma unroll
    for (int f = 0; f < 4; ++f)
#pragma unroll
      for (int e = 0; e < 8; ++e) {
        int m = mbase + f * 16 + mrow + e;
        int g = m >> 5, j = m & 31;
        u[((size_t)b * 1152 + (j * 36 + s)) * 8 + g] = acc[nsub * 4 + f][e] + pbias[m];
      }
  }
}

// ---------------- squash over the 8-dim capsule axis ------------------------

__global__ void squash_u(float* __restrict__ u, int n) {
  int i = blockIdx.x * 256 + threadIdx.x;
  if (i >= n) return;
  float* p = u + (size_t)i * 8;
  float vv[8], sq = 0.f;
#pragma unroll
  for (int g = 0; g < 8; ++g) { vv[g] = p[g]; sq += vv[g] * vv[g]; }
  float scale = (sq / (1.f + sq)) / fmaxf(sqrtf(sq), 1e-20f);
#pragma unroll
  for (int g = 0; g < 8; ++g) p[g] = vv[g] * scale;
}

// ---------------- routing: softmax over r (block per (c,b)) -----------------

__global__ void softmax_r(const float* __restrict__ logits, float* __restrict__ probs) {
  int cb = blockIdx.x;
  const float* L = logits + (size_t)cb * 1152;
  float*       P = probs  + (size_t)cb * 1152;
  __shared__ float red[256];
  int t = threadIdx.x;
  float mx = -1e30f;
  for (int r = t; r < 1152; r += 256) mx = fmaxf(mx, L[r]);
  red[t] = mx; __syncthreads();
  for (int s = 128; s > 0; s >>= 1) {
    if (t < s) red[t] = fmaxf(red[t], red[t + s]);
    __syncthreads();
  }
  mx = red[0]; __syncthreads();
  float sm = 0.f;
  for (int r = t; r < 1152; r += 256) sm += expf(L[r] - mx);
  red[t] = sm; __syncthreads();
  for (int s = 128; s > 0; s >>= 1) {
    if (t < s) red[t] += red[t + s];
    __syncthreads();
  }
  float inv = 1.f / red[0];
  for (int r = t; r < 1152; r += 256) P[r] = expf(L[r] - mx) * inv;
}

// ------ s = sum_r probs*prior (priors recomputed on the fly); v=squash ------

__global__ void accum_v(const float* __restrict__ u, const float* __restrict__ rw,
                        const float* __restrict__ probs, float* __restrict__ v) {
  int cb = blockIdx.x; int c = cb / BATCH; int b = cb % BATCH;
  float acc[16];
#pragma unroll
  for (int o = 0; o < 16; ++o) acc[o] = 0.f;
  for (int r = threadIdx.x; r < 1152; r += 256) {
    const float* up = u  + ((size_t)b * 1152 + r) * 8;
    const float* wp = rw + ((size_t)c * 1152 + r) * 128;
    float pr = probs[(size_t)cb * 1152 + r];
    float ui[8];
#pragma unroll
    for (int i = 0; i < 8; ++i) ui[i] = up[i] * pr;
#pragma unroll
    for (int o = 0; o < 16; ++o) {
      float tt = 0.f;
#pragma unroll
      for (int i = 0; i < 8; ++i) tt = fmaf(ui[i], wp[i * 16 + o], tt);
      acc[o] += tt;
    }
  }
  __shared__ float red[256 * 16];
#pragma unroll
  for (int o = 0; o < 16; ++o) red[threadIdx.x * 16 + o] = acc[o];
  __syncthreads();
  for (int s = 128; s > 0; s >>= 1) {
    if (threadIdx.x < s)
#pragma unroll
      for (int o = 0; o < 16; ++o)
        red[threadIdx.x * 16 + o] += red[(threadIdx.x + s) * 16 + o];
    __syncthreads();
  }
  if (threadIdx.x == 0) {
    float sq = 0.f;
#pragma unroll
    for (int o = 0; o < 16; ++o) sq += red[o] * red[o];
    float scale = (sq / (1.f + sq)) / fmaxf(sqrtf(sq), 1e-20f);
#pragma unroll
    for (int o = 0; o < 16; ++o) v[(size_t)cb * 16 + o] = red[o] * scale;
  }
}

// ---------------- logits += sum_o prior*v -----------------------------------

__global__ void update_logits(const float* __restrict__ u, const float* __restrict__ rw,
                              const float* __restrict__ v, float* __restrict__ logits) {
  int i = blockIdx.x * 256 + threadIdx.x;            // [c][b][r]
  if (i >= NUM_CLASSES * BATCH * 1152) return;
  int r = i % 1152; int t = i / 1152; int b = t % BATCH; int c = t / BATCH;
  const float* up = u  + ((size_t)b * 1152 + r) * 8;
  const float* wp = rw + ((size_t)c * 1152 + r) * 128;
  const float* vp = v  + ((size_t)c * BATCH + b) * 16;
  float dot = 0.f;
#pragma unroll
  for (int o = 0; o < 16; ++o) {
    float pr = 0.f;
#pragma unroll
    for (int k = 0; k < 8; ++k) pr = fmaf(up[k], wp[k * 16 + o], pr);
    dot = fmaf(pr, vp[o], dot);
  }
  logits[i] += dot;
}

// -------- class lengths, softmax, argmax mask; caps -> bf16 [b][160] --------

__global__ void classes_mask(const float* __restrict__ v, float* __restrict__ out_classes,
                             __bf16* __restrict__ caps) {
  int b = blockIdx.x;
  int t = threadIdx.x;                     // 32 threads (one wave)
  __shared__ float len[NUM_CLASSES], cls[NUM_CLASSES];
  __shared__ int amax;
  if (t < NUM_CLASSES) {
    float sq = 0.f;
#pragma unroll
    for (int o = 0; o < 16; ++o) {
      float x = v[((size_t)t * BATCH + b) * 16 + o];
      sq += x * x;
    }
    len[t] = sqrtf(sq);
  }
  __syncthreads();
  if (t == 0) {
    float mx = len[0]; int am = 0;
    for (int c = 1; c < NUM_CLASSES; ++c) if (len[c] > mx) { mx = len[c]; am = c; }
    float sm = 0.f;
    for (int c = 0; c < NUM_CLASSES; ++c) sm += expf(len[c] - mx);
    for (int c = 0; c < NUM_CLASSES; ++c) cls[c] = expf(len[c] - mx) / sm;
    amax = am;
  }
  __syncthreads();
  if (t < NUM_CLASSES) out_classes[(size_t)b * NUM_CLASSES + t] = cls[t];
  for (int i = t; i < 160; i += 32) {
    int c = i >> 4, o = i & 15;
    float val = (c == amax) ? v[((size_t)c * BATCH + b) * 16 + o] : 0.f;
    caps[(size_t)b * 160 + i] = (__bf16)val;
  }
}

// -------- decoder GEMM: out[M=512,N] = act(A[M,K] @ W[N,K]^T + bias) --------
// ACT 0: relu -> bf16 activations; ACT 1: sigmoid -> f32 (final recon)

template <int ACT>
__global__ void gemm_dec(const __bf16* __restrict__ A, const __bf16* __restrict__ W,
                         const float* __restrict__ bias, void* __restrict__ out,
                         int K, int N) {
  int lane   = threadIdx.x & 31;
  int wid    = blockIdx.x * 8 + (threadIdx.x >> 5);
  int ntiles = N >> 4;
  int mbase  = (wid / ntiles) * 64;
  int nbase  = (wid % ntiles) * 16;
  int nn     = nbase + (lane & 15);
  int khalfA = (lane < 16) ? 0 : 8;
  int khalfB = (lane < 16) ? 0 : 16;

  v8f zero = {0.f, 0.f, 0.f, 0.f, 0.f, 0.f, 0.f, 0.f};
  v8f acc[4] = {zero, zero, zero, zero};

  for (int k0 = 0; k0 < K; k0 += 32) {
    v16bf bf = *(const v16bf*)(W + (size_t)nn * K + k0 + khalfB);
#pragma unroll
    for (int f = 0; f < 4; ++f) {
      v16bf af = load_a_frag(A + (size_t)(mbase + f * 16 + (lane & 15)) * K + k0, khalfA);
      acc[f] = __builtin_amdgcn_wmma_f32_16x16x32_bf16(false, af, false, bf,
                                                       (short)0, acc[f], false, false);
    }
  }

  float bs = bias[nn];
  int mrow = (lane < 16) ? 0 : 8;
#pragma unroll
  for (int f = 0; f < 4; ++f)
#pragma unroll
    for (int e = 0; e < 8; ++e) {
      int m = mbase + f * 16 + mrow + e;
      float val = acc[f][e] + bs;
      if (ACT == 0) {
        ((__bf16*)out)[(size_t)m * N + nn] = (__bf16)fmaxf(val, 0.f);
      } else {
        ((float*)out)[(size_t)m * N + nn] = 1.f / (1.f + expf(-val));
      }
    }
}

// ---------------------------------------------------------------------------

extern "C" void kernel_launch(void* const* d_in, const int* in_sizes, int n_in,
                              void* d_out, int out_size, void* d_ws, size_t ws_size,
                              hipStream_t stream) {
  (void)in_sizes; (void)n_in; (void)out_size;
  const float* x       = (const float*)d_in[0];
  const float* conv1_w = (const float*)d_in[1];
  const float* conv1_b = (const float*)d_in[2];
  const float* prim_w  = (const float*)d_in[3];
  const float* prim_b  = (const float*)d_in[4];
  const float* route_w = (const float*)d_in[5];
  const float* dec_w1  = (const float*)d_in[6];
  const float* dec_b1  = (const float*)d_in[7];
  const float* dec_w2  = (const float*)d_in[8];
  const float* dec_b2  = (const float*)d_in[9];
  const float* dec_w3  = (const float*)d_in[10];
  const float* dec_b3  = (const float*)d_in[11];

  // workspace layout (bytes, all 256-aligned)
  char* ws = (char*)d_ws;
  const size_t OFF_HT    = 0;                               // 512*400*256 bf16
  const size_t OFF_WA    = OFF_HT    + 104857600;           // 81*256*256 bf16
  const size_t OFF_U     = OFF_WA    + 10616832;            // 512*1152*8 f32
  const size_t OFF_LOG   = OFF_U     + 18874368;            // 10*512*1152 f32
  const size_t OFF_PROB  = OFF_LOG   + 23592960;            // 10*512*1152 f32
  const size_t OFF_V     = OFF_PROB  + 23592960;            // 10*512*16 f32
  const size_t OFF_CAPS  = OFF_V     + 327680;              // 512*160 bf16
  const size_t OFF_W1B   = OFF_CAPS  + 163840;              // 512*160 bf16
  const size_t OFF_W2B   = OFF_W1B   + 163840;              // 1024*512 bf16
  const size_t OFF_W3B   = OFF_W2B   + 1048576;             // 784*1024 bf16
  const size_t OFF_D1    = OFF_W3B   + 1605632;             // 512*512 bf16
  const size_t OFF_D2    = OFF_D1    + 524288;              // 512*1024 bf16
  const size_t TOTAL     = OFF_D2    + 1048576;
  if (ws_size < TOTAL) return;

  __bf16* hT     = (__bf16*)(ws + OFF_HT);
  __bf16* wA     = (__bf16*)(ws + OFF_WA);
  float*  u      = (float*) (ws + OFF_U);
  float*  logits = (float*) (ws + OFF_LOG);
  float*  probs  = (float*) (ws + OFF_PROB);
  float*  v      = (float*) (ws + OFF_V);
  __bf16* caps   = (__bf16*)(ws + OFF_CAPS);
  __bf16* w1b    = (__bf16*)(ws + OFF_W1B);
  __bf16* w2b    = (__bf16*)(ws + OFF_W2B);
  __bf16* w3b    = (__bf16*)(ws + OFF_W3B);
  __bf16* d1     = (__bf16*)(ws + OFF_D1);
  __bf16* d2     = (__bf16*)(ws + OFF_D2);

  float* out_classes = (float*)d_out;                       // [512,10]
  float* out_recon   = (float*)d_out + BATCH * NUM_CLASSES; // [512,784]

  // stage 0: conversions / packing
  conv1_kernel<<<BATCH * 400, 256, 0, stream>>>(x, conv1_w, conv1_b, hT);
  prep_wA<<<(5308416 + 255) / 256, 256, 0, stream>>>(prim_w, wA, 5308416);
  f32_to_bf16<<<(81920   + 255) / 256, 256, 0, stream>>>(dec_w1, w1b, 81920);
  f32_to_bf16<<<(524288  + 255) / 256, 256, 0, stream>>>(dec_w2, w2b, 524288);
  f32_to_bf16<<<(802816  + 255) / 256, 256, 0, stream>>>(dec_w3, w3b, 802816);

  // stage 1: primary capsules (the 196-GFLOP WMMA GEMM) + squash
  prim_gemm<<<288, 256, 0, stream>>>(hT, wA, prim_b, u);
  squash_u<<<(589824 + 255) / 256, 256, 0, stream>>>(u, 589824);

  // stage 2: dynamic routing (3 iterations, priors recomputed on the fly)
  const int nlog = NUM_CLASSES * BATCH * 1152;
  fill_zero<<<(nlog + 255) / 256, 256, 0, stream>>>(logits, nlog);
  for (int it = 0; it < 3; ++it) {
    softmax_r<<<NUM_CLASSES * BATCH, 256, 0, stream>>>(logits, probs);
    accum_v<<<NUM_CLASSES * BATCH, 256, 0, stream>>>(u, route_w, probs, v);
    if (it < 2)
      update_logits<<<(nlog + 255) / 256, 256, 0, stream>>>(u, route_w, v, logits);
  }

  // stage 3: class probabilities + argmax mask
  classes_mask<<<BATCH, 32, 0, stream>>>(v, out_classes, caps);

  // stage 4: decoder MLP (WMMA)
  gemm_dec<0><<<32, 256, 0, stream>>>(caps, w1b, dec_b1, d1, 160, 512);
  gemm_dec<0><<<64, 256, 0, stream>>>(d1,   w2b, dec_b2, d2, 512, 1024);
  gemm_dec<1><<<49, 256, 0, stream>>>(d2,   w3b, dec_b3, out_recon, 1024, 784);
}
